// Inforpatch_addfusion_81295140979222
// MI455X (gfx1250) — matrix operand order repaired and checked by reading, not compiled, not run on hardware
//
#include <hip/hip_runtime.h>
#include <hip/hip_bf16.h>

#define CDIM 128
#define HA 144
#define WA 176
#define HC 128
#define WC 160
#define PH 8
#define PW 8
#define HWC (HC*WC)      // 20480
#define HAWA (HA*WA)     // 25344

typedef __attribute__((ext_vector_type(16))) __bf16 v16bf;
typedef __attribute__((ext_vector_type(8)))  float  v8f;

union ABReg { v16bf v; unsigned u[8]; };

// ---------------------------------------------------------------------------
// Kernel 1: per-branch 1x1 conv (channel GEMM via bf16 WMMA) + channel mean/max
// grid: (40, B, branch), block 256 (8 waves). Each wave: 4 tiles of 16 pixels.
// s layout: s[((b*2+br)*2 + plane)*HWC + pix], plane 0 = mean, 1 = max
// ---------------------------------------------------------------------------
__global__ void __launch_bounds__(256)
branch_conv_kernel(const float* __restrict__ x1, const float* __restrict__ x2,
                   const float* __restrict__ w1, const float* __restrict__ b1,
                   const float* __restrict__ w2, const float* __restrict__ b2,
                   float* __restrict__ sOut)
{
  __shared__ __align__(16) __bf16 Wlds[CDIM*CDIM];
  __shared__ __align__(16) float biasLds[CDIM];

  const int br = blockIdx.z;
  const int b  = blockIdx.y;
  const float* __restrict__ xsrc = (br == 0) ? x1 : x2;
  const float* __restrict__ wsrc = (br == 0) ? w1 : w2;
  const float* __restrict__ bsrc = (br == 0) ? b1 : b2;

  // Stage weights as bf16 (native v_cvt) and bias as f32
  for (int i = threadIdx.x; i < CDIM*CDIM; i += 256) Wlds[i] = (__bf16)wsrc[i];
  if (threadIdx.x < CDIM) biasLds[threadIdx.x] = bsrc[threadIdx.x];
  __syncthreads();

  const int lane = threadIdx.x & 31;
  const int wv   = threadIdx.x >> 5;
  const int n    = lane & 15;   // pixel column within tile / matrix row M
  const int g    = lane >> 4;   // lane half

  for (int t = 0; t < 4; ++t) {
    const int tile = (blockIdx.x * 8 + wv) * 4 + t;   // 0..1279
    const int pix0 = tile * 16;                       // 16 consecutive pixels, same row
    const int h    = pix0 / WC;
    const int w0   = pix0 % WC;
    const float* __restrict__ gbase =
        xsrc + (size_t)b * CDIM * HAWA + (size_t)(PH + h) * WA + (PW + w0 + n);

    v8f acc[8];
    #pragma unroll
    for (int m = 0; m < 8; ++m)
      #pragma unroll
      for (int j = 0; j < 8; ++j) acc[m][j] = 0.0f;

    for (int k0 = 0; k0 < CDIM; k0 += 32) {
      // B operand: X[c][pix], 32x16 bf16. lane half g holds K = g*16 + 0..15.
      ABReg bt;
      #pragma unroll
      for (int v = 0; v < 8; ++v) {
        const int c = k0 + g*16 + 2*v;
        bt.v[2*v]   = (__bf16)gbase[(size_t)c       * HAWA];
        bt.v[2*v+1] = (__bf16)gbase[(size_t)(c + 1) * HAWA];
      }
      // A operand: W[o][c], 16x32 bf16: VGPR0-3 K=g*16+0..7,
      // VGPR4-7 K=g*16+8..15 -> two contiguous 16B LDS reads per lane.
      #pragma unroll
      for (int m = 0; m < 8; ++m) {
        const int o = m*16 + n;
        const uint4 q0 = *(const uint4*)&Wlds[o*CDIM + k0 + g*16];
        const uint4 q1 = *(const uint4*)&Wlds[o*CDIM + k0 + g*16 + 8];
        ABReg at;
        at.u[0]=q0.x; at.u[1]=q0.y; at.u[2]=q0.z; at.u[3]=q0.w;
        at.u[4]=q1.x; at.u[5]=q1.y; at.u[6]=q1.z; at.u[7]=q1.w;
        acc[m] = __builtin_amdgcn_wmma_f32_16x16x32_bf16(
                     false, at.v, false, bt.v, (short)0, acc[m], false, false);
      }
    }

    // C/D layout: acc[m][j] = channel (m*16 + g*8 + j), pixel n. Add bias,
    // reduce mean/max over channels; combine lane halves with shfl_xor(16).
    float sum = 0.0f, vmax = -3.4e38f;
    #pragma unroll
    for (int m = 0; m < 8; ++m) {
      const float4 bA = *(const float4*)&biasLds[m*16 + g*8];
      const float4 bB = *(const float4*)&biasLds[m*16 + g*8 + 4];
      const float bv[8] = {bA.x, bA.y, bA.z, bA.w, bB.x, bB.y, bB.z, bB.w};
      #pragma unroll
      for (int j = 0; j < 8; ++j) {
        const float val = acc[m][j] + bv[j];
        sum += val;
        vmax = fmaxf(vmax, val);
      }
    }
    sum  += __shfl_xor(sum, 16);
    vmax  = fmaxf(vmax, __shfl_xor(vmax, 16));
    if (lane < 16) {
      const size_t base = (size_t)((b*2 + br) * 2) * HWC + pix0 + n;
      sOut[base]       = sum * (1.0f / 128.0f);
      sOut[base + HWC] = vmax;
    }
  }
}

// ---------------------------------------------------------------------------
// Kernel 2: 7x7 conv over the 2-channel (mean,max) map, zero padding (3,3)
// grid: (80, B, branch), 256 threads -> one thread per pixel
// ---------------------------------------------------------------------------
__global__ void __launch_bounds__(256)
conv7_kernel(const float* __restrict__ sIn, const float* __restrict__ cw1,
             const float* __restrict__ cw2, float* __restrict__ x12)
{
  __shared__ float wt[98];
  const int br = blockIdx.z, b = blockIdx.y;
  const float* __restrict__ cw = (br == 0) ? cw1 : cw2;
  if (threadIdx.x < 98) wt[threadIdx.x] = cw[threadIdx.x];
  __syncthreads();

  const int pix = blockIdx.x * 256 + threadIdx.x;   // < 20480
  const int h = pix / WC, w = pix % WC;
  const float* __restrict__ sb = sIn + (size_t)(b*2 + br) * 2 * HWC;

  float acc = 0.0f;
  #pragma unroll
  for (int c2 = 0; c2 < 2; ++c2) {
    const float* __restrict__ sp = sb + c2 * HWC;
    for (int kh = 0; kh < 7; ++kh) {
      const int hh = h + kh - 3;
      if (hh < 0 || hh >= HC) continue;
      for (int kw = 0; kw < 7; ++kw) {
        const int ww = w + kw - 3;
        if (ww < 0 || ww >= WC) continue;
        acc += wt[c2*49 + kh*7 + kw] * sp[hh*WC + ww];
      }
    }
  }
  x12[(size_t)(b*2 + br) * HWC + pix] = acc;
}

// ---------------------------------------------------------------------------
// Kernel 3: per 8x8 patch: std(ddof=1) + reflect-pad gradient mean, softmax
// over branches -> w1p[b*320 + patch]. One wave per (b,patch).
// grid: (320), 256 threads (8 waves)
// ---------------------------------------------------------------------------
__global__ void __launch_bounds__(256)
patch_stats_kernel(const float* __restrict__ x12, float* __restrict__ w1p)
{
  __shared__ float wbuf[8][2][64];
  const int lane = threadIdx.x & 31, wv = threadIdx.x >> 5;
  const int gp = blockIdx.x * 8 + wv;     // 0..2559
  const int b  = gp / 320, pp = gp % 320;
  const int pi = pp / 20, pj = pp % 20;

  float sum[2], sumsq[2], gsum[2];

  #pragma unroll
  for (int br = 0; br < 2; ++br) {
    const float* __restrict__ src =
        x12 + (size_t)(b*2 + br) * HWC + (pi*8)*WC + pj*8;
    const int i0 = lane, i1 = lane + 32;
    const float v0 = src[(i0 >> 3) * WC + (i0 & 7)];
    const float v1 = src[(i1 >> 3) * WC + (i1 & 7)];
    wbuf[wv][br][i0] = v0;
    wbuf[wv][br][i1] = v1;
    float s = v0 + v1, q = v0*v0 + v1*v1;
    #pragma unroll
    for (int off = 16; off; off >>= 1) {
      s += __shfl_xor(s, off);
      q += __shfl_xor(q, off);
    }
    sum[br] = s; sumsq[br] = q;
  }
  __syncthreads();

  #pragma unroll
  for (int br = 0; br < 2; ++br) {
    const float* __restrict__ wp = &wbuf[wv][br][0];
    float gacc = 0.0f;
    #pragma unroll
    for (int t = 0; t < 2; ++t) {
      const int idx = lane + t*32;
      const int r = idx >> 3, c = idx & 7;
      // reflect pad-1 within the 8x8 window: -1 -> 1, 8 -> 6
      auto WV = [&](int a, int bb) -> float {
        const int aa = a  < 0 ? -a  : (a  > 7 ? 14 - a  : a);
        const int bc = bb < 0 ? -bb : (bb > 7 ? 14 - bb : bb);
        return wp[aa*8 + bc];
      };
      const float gx = -WV(r-1,c-1) + WV(r-1,c+1)
                       -2.0f*WV(r,c-1) + 2.0f*WV(r,c+1)
                       -WV(r+1,c-1) + WV(r+1,c+1);
      const float gy = -WV(r-1,c-1) - 2.0f*WV(r-1,c) - WV(r-1,c+1)
                       +WV(r+1,c-1) + 2.0f*WV(r+1,c) + WV(r+1,c+1);
      gacc += fabsf(gx) + fabsf(gy);
    }
    #pragma unroll
    for (int off = 16; off; off >>= 1) gacc += __shfl_xor(gacc, off);
    gsum[br] = gacc * (1.0f / 64.0f);
  }

  const float std0 = sqrtf(fmaxf(0.0f, (sumsq[0] - sum[0]*sum[0]*(1.0f/64.0f)) * (1.0f/63.0f)));
  const float std1 = sqrtf(fmaxf(0.0f, (sumsq[1] - sum[1]*sum[1]*(1.0f/64.0f)) * (1.0f/63.0f)));
  const float e0 = std0 + gsum[0];
  const float e1 = std1 + gsum[1];
  const float w1 = 1.0f / (1.0f + __expf(e1 - e0));   // softmax over 2
  if (lane == 0) w1p[gp] = w1;
}

// ---------------------------------------------------------------------------
// Kernel 4: blend with reflect-padded, 8x-upsampled patch weights (float4 I/O)
// grid: (3168, B), 256 threads; 4 f32 per thread
// ---------------------------------------------------------------------------
__global__ void __launch_bounds__(256)
blend_kernel(const float* __restrict__ x1, const float* __restrict__ x2,
             const float* __restrict__ w1p, float* __restrict__ out)
{
  __shared__ float wl[320];
  const int b = blockIdx.y;
  for (int i = threadIdx.x; i < 320; i += 256) wl[i] = w1p[b*320 + i];
  __syncthreads();

  const size_t e0   = ((size_t)blockIdx.x * 256 + threadIdx.x) * 4;  // < CDIM*HAWA
  const size_t base = (size_t)b * CDIM * HAWA + e0;
  const float4 a = *(const float4*)(x1 + base);
  const float4 c = *(const float4*)(x2 + base);

  const int rem = (int)(e0 % HAWA);
  const int hh  = rem / WA;
  const int ww0 = rem % WA;
  int h = hh - PH;
  h = (h < 0) ? -h : ((h > HC-1) ? 2*(HC-1) - h : h);    // reflect into [0,128)
  const int prow = (h >> 3) * 20;

  const float av[4] = {a.x, a.y, a.z, a.w};
  const float cv[4] = {c.x, c.y, c.z, c.w};
  float ov[4];
  #pragma unroll
  for (int k = 0; k < 4; ++k) {
    int w = ww0 + k - PW;
    w = (w < 0) ? -w : ((w > WC-1) ? 2*(WC-1) - w : w);  // reflect into [0,160)
    const float wt = wl[prow + (w >> 3)];
    ov[k] = av[k]*wt + cv[k]*(1.0f - wt);
  }
  float4 o; o.x = ov[0]; o.y = ov[1]; o.z = ov[2]; o.w = ov[3];
  *(float4*)(out + base) = o;
}

// ---------------------------------------------------------------------------
extern "C" void kernel_launch(void* const* d_in, const int* in_sizes, int n_in,
                              void* d_out, int out_size, void* d_ws, size_t ws_size,
                              hipStream_t stream) {
  const float* x1      = (const float*)d_in[0];
  const float* x2      = (const float*)d_in[1];
  const float* conv1_w = (const float*)d_in[2];
  const float* conv1_b = (const float*)d_in[3];
  const float* cv1_w   = (const float*)d_in[4];
  const float* conv2_w = (const float*)d_in[5];
  const float* conv2_b = (const float*)d_in[6];
  const float* cv2_w   = (const float*)d_in[7];
  float* out = (float*)d_out;

  // Workspace layout (floats): s[8*2*2*20480] | x12[8*2*20480] | w1p[8*320]
  float* s   = (float*)d_ws;
  float* x12 = s   + (size_t)8*2*2*HWC;
  float* w1p = x12 + (size_t)8*2*HWC;

  branch_conv_kernel<<<dim3(40, 8, 2), 256, 0, stream>>>(
      x1, x2, conv1_w, conv1_b, conv2_w, conv2_b, s);
  conv7_kernel<<<dim3(80, 8, 2), 256, 0, stream>>>(s, cv1_w, cv2_w, x12);
  patch_stats_kernel<<<dim3(320), 256, 0, stream>>>(x12, w1p);
  blend_kernel<<<dim3(3168, 8), 256, 0, stream>>>(x1, x2, w1p, out);
}
